// Attention_31988916420764
// MI455X (gfx1250) — matrix operand order repaired and checked by reading, not compiled, not run on hardware
//
#include <hip/hip_runtime.h>

#define D_MODEL 1024
#define S_LEN   4096
#define B_SZ    4

typedef __bf16 bf16_t;
typedef bf16_t v16bf __attribute__((ext_vector_type(16)));
typedef bf16_t v8bf  __attribute__((ext_vector_type(8)));
typedef bf16_t v4bf  __attribute__((ext_vector_type(4)));
typedef float  v8f   __attribute__((ext_vector_type(8)));

union FragBF16 {
    v16bf v;
    v8bf  h[2];
};

// Async copy of 32 bytes global -> LDS (2 x b128). The 24-bit instruction
// offset applies to BOTH the LDS destination and the global source address
// (CDNA5 ISA), so one base pair covers both chunks.
#define ASYNC_CP32(lds_off, gptr)                                          \
    asm volatile("global_load_async_to_lds_b128 %0, %1, off\n\t"           \
                 "global_load_async_to_lds_b128 %0, %1, off offset:16"     \
                 :: "v"(lds_off), "v"(gptr) : "memory")

#define WAIT_ASYNC0() asm volatile("s_wait_asynccnt 0x0" ::: "memory")

// ---------------------------------------------------------------------------
// Kernel 1: fused dual GEMM.
//   H = hidden @ w1 + b1             (bf16 out)
//   I = (hidden + pre_emb) @ w2 + b2 (bf16 out)
// Workgroup tile: 64 rows x 64 cols. 8 waves; wave = 16 rows x 32 cols,
// computing both H and I tiles (shares the staged hidden tile).
// fp32 -> bf16 conversion must pass through VGPRs, so this kernel keeps the
// packed load/convert/ds_store staging path.
// ---------------------------------------------------------------------------
__global__ __launch_bounds__(256) void fused_linear_kernel(
    const float* __restrict__ hidden, const float* __restrict__ pre_emb,
    const float* __restrict__ w1, const float* __restrict__ b1,
    const float* __restrict__ w2, const float* __restrict__ b2,
    bf16_t* __restrict__ Hout, bf16_t* __restrict__ Iout)
{
    __shared__ __align__(16) bf16_t lA [64][32];   // hidden tile (bf16)
    __shared__ __align__(16) bf16_t lAI[64][32];   // hidden + pre_emb tile
    __shared__ __align__(16) bf16_t lW1[64][32];   // w1 tile, stored [n][k]
    __shared__ __align__(16) bf16_t lW2[64][32];   // w2 tile, stored [n][k]

    const int tid    = threadIdx.x;
    const int lane   = tid & 31;
    const int wave   = tid >> 5;
    const int l      = lane & 15;      // N / M index within 16
    const int hi     = lane >> 4;      // K-half / M-half selector
    const int m0     = blockIdx.y * 64;
    const int n0     = blockIdx.x * 64;
    const int rowBlk = (wave & 3) * 16;    // wave's row offset in tile
    const int colBlk = (wave >> 2) * 32;   // wave's col offset in tile

    // staging coordinates (hoisted out of the K loop)
    const int ar = tid >> 2;            // 0..63   A-tile row
    const int ac = (tid & 3) * 8;       // 0,8,16,24
    const int n2 = (tid & 31) * 2;      // 0..62   W-tile n pair (coalesced)
    const int kq = (tid >> 5) * 4;      // 0..28   W-tile k quad

    v8f accH[2] = {};
    v8f accI[2] = {};

    for (int k0 = 0; k0 < D_MODEL; k0 += 32) {
        // --- stage A tiles: 64 rows x 32 K, fp32 -> bf16, packed b128 store ---
        {
            const float4* hp4 = (const float4*)(hidden  + (size_t)(m0 + ar) * D_MODEL + k0 + ac);
            const float4* pp4 = (const float4*)(pre_emb + (size_t)(m0 + ar) * D_MODEL + k0 + ac);
            const float4 h0 = hp4[0], h1 = hp4[1];
            const float4 p0 = pp4[0], p1 = pp4[1];
            const float hs[8] = {h0.x, h0.y, h0.z, h0.w, h1.x, h1.y, h1.z, h1.w};
            const float ps[8] = {p0.x, p0.y, p0.z, p0.w, p1.x, p1.y, p1.z, p1.w};
            v8bf va, vb;
#pragma unroll
            for (int j = 0; j < 8; ++j) {
                va[j] = (bf16_t)hs[j];
                vb[j] = (bf16_t)(hs[j] + ps[j]);
            }
            *(v8bf*)&lA [ar][ac] = va;
            *(v8bf*)&lAI[ar][ac] = vb;
        }
        // --- stage W tiles transposed to [n][k]: 2(n) x 4(k) micro-tile per
        //     thread, coalesced float2 global loads, packed b64 LDS stores ---
        {
            const float2* w1p = (const float2*)(w1 + (size_t)(k0 + kq) * D_MODEL + n0 + n2);
            const float2* w2p = (const float2*)(w2 + (size_t)(k0 + kq) * D_MODEL + n0 + n2);
            v4bf q1a, q1b, q2a, q2b;
#pragma unroll
            for (int j = 0; j < 4; ++j) {
                const float2 a = w1p[j * (D_MODEL / 2)];
                const float2 b = w2p[j * (D_MODEL / 2)];
                q1a[j] = (bf16_t)a.x;  q1b[j] = (bf16_t)a.y;
                q2a[j] = (bf16_t)b.x;  q2b[j] = (bf16_t)b.y;
            }
            *(v4bf*)&lW1[n2][kq]     = q1a;
            *(v4bf*)&lW1[n2 + 1][kq] = q1b;
            *(v4bf*)&lW2[n2][kq]     = q2a;
            *(v4bf*)&lW2[n2 + 1][kq] = q2b;
        }
        __syncthreads();

        // --- A fragments (16x32 bf16 layout: lanes<16 K=0..7,16..23) ---
        FragBF16 aH, aI;
        const bf16_t* ra = &lA[rowBlk + l][0];
        aH.h[0] = *(const v8bf*)(ra + hi * 8);
        aH.h[1] = *(const v8bf*)(ra + 16 + hi * 8);
        const bf16_t* ri = &lAI[rowBlk + l][0];
        aI.h[0] = *(const v8bf*)(ri + hi * 8);
        aI.h[1] = *(const v8bf*)(ri + 16 + hi * 8);

#pragma unroll
        for (int nt = 0; nt < 2; ++nt) {
            // B fragments: lane holds column N, K-contiguous in LDS row
            const bf16_t* p1 = &lW1[colBlk + nt * 16 + l][0] + hi * 8;
            FragBF16 bW1; bW1.h[0] = *(const v8bf*)p1; bW1.h[1] = *(const v8bf*)(p1 + 16);
            const bf16_t* p2 = &lW2[colBlk + nt * 16 + l][0] + hi * 8;
            FragBF16 bW2; bW2.h[0] = *(const v8bf*)p2; bW2.h[1] = *(const v8bf*)(p2 + 16);

            accH[nt] = __builtin_amdgcn_wmma_f32_16x16x32_bf16(
                false, aH.v, false, bW1.v, (short)0, accH[nt], false, false);
            accI[nt] = __builtin_amdgcn_wmma_f32_16x16x32_bf16(
                false, aI.v, false, bW2.v, (short)0, accI[nt], false, false);
        }
        __syncthreads();
    }

    // --- epilogue: add bias, convert to bf16, store ---
#pragma unroll
    for (int nt = 0; nt < 2; ++nt) {
        const int col = n0 + colBlk + nt * 16 + l;
        const float bias1 = b1[col];
        const float bias2 = b2[col];
#pragma unroll
        for (int r = 0; r < 8; ++r) {
            const int row = m0 + rowBlk + hi * 8 + r;
            Hout[(size_t)row * D_MODEL + col] = (bf16_t)(accH[nt][r] + bias1);
            Iout[(size_t)row * D_MODEL + col] = (bf16_t)(accI[nt][r] + bias2);
        }
    }
}

// ---------------------------------------------------------------------------
// Kernel 2: attn[b,q,k] = sum_d H[b,q,d] * I[b,k,d]   (H @ I^T), fp32 out.
// Workgroup tile: 128q x 128k. 8 waves as 2(q) x 4(k); wave = 64q x 32k
// = 4x2 WMMA tiles. Double-buffered LDS tiles filled by
// GLOBAL_LOAD_ASYNC_TO_LDS_B128 (ASYNCcnt), overlapping copy of tile t+1
// with WMMA on tile t; one barrier per K-step.
// ---------------------------------------------------------------------------
__global__ __launch_bounds__(256) void attn_scores_kernel(
    const bf16_t* __restrict__ H, const bf16_t* __restrict__ I,
    float* __restrict__ out)
{
    __shared__ __align__(16) bf16_t lH[2][128][32];
    __shared__ __align__(16) bf16_t lI[2][128][32];

    const int tid  = threadIdx.x;
    const int lane = tid & 31;
    const int wave = tid >> 5;
    const int l    = lane & 15;
    const int hi   = lane >> 4;
    const int b    = blockIdx.z;
    const int q0   = blockIdx.y * 128;
    const int k0   = blockIdx.x * 128;
    const int qw   = (wave >> 2) * 64;   // wave q offset (0 or 64)
    const int kw   = (wave & 3) * 32;    // wave k offset (0,32,64,96)

    const bf16_t* Hb = H + (size_t)b * S_LEN * D_MODEL;
    const bf16_t* Ib = I + (size_t)b * S_LEN * D_MODEL;

    const int sr = tid >> 1;             // staging row 0..127
    const int sc = (tid & 1) * 16;       // staging col 0 or 16

    // per-thread LDS byte offsets (wave-relative) for the async destination
    const unsigned ldsH = (unsigned)(size_t)&lH[0][sr][sc];
    const unsigned ldsI = (unsigned)(size_t)&lI[0][sr][sc];
    const unsigned bufStride = 128u * 32u * (unsigned)sizeof(bf16_t);  // 8 KB

    const bf16_t* hp = Hb + (size_t)(q0 + sr) * D_MODEL + sc;
    const bf16_t* ip = Ib + (size_t)(k0 + sr) * D_MODEL + sc;

    v8f acc[4][2] = {};

    // prefetch tile 0 into buffer 0
    ASYNC_CP32(ldsH, hp);
    ASYNC_CP32(ldsI, ip);

    const int NSTEP = D_MODEL / 32;
    for (int it = 0; it < NSTEP; ++it) {
        // my async writes done; barrier => everyone's writes visible
        WAIT_ASYNC0();
        __syncthreads();

        const int cur = it & 1;
        if (it + 1 < NSTEP) {
            const unsigned nb = (unsigned)(cur ^ 1) * bufStride;
            ASYNC_CP32(ldsH + nb, hp + (it + 1) * 32);
            ASYNC_CP32(ldsI + nb, ip + (it + 1) * 32);
        }

        const bf16_t (*bH)[32] = lH[cur];
        const bf16_t (*bI)[32] = lI[cur];

        // B fragments: lane = column n (attention-k index), d contiguous
        FragBF16 bfr[2];
#pragma unroll
        for (int nt = 0; nt < 2; ++nt) {
            const bf16_t* p = &bI[kw + nt * 16 + l][0] + hi * 8;
            bfr[nt].h[0] = *(const v8bf*)p;
            bfr[nt].h[1] = *(const v8bf*)(p + 16);
        }
#pragma unroll
        for (int mt = 0; mt < 4; ++mt) {
            const bf16_t* ra = &bH[qw + mt * 16 + l][0];
            FragBF16 a;
            a.h[0] = *(const v8bf*)(ra + hi * 8);
            a.h[1] = *(const v8bf*)(ra + 16 + hi * 8);
#pragma unroll
            for (int nt = 0; nt < 2; ++nt) {
                acc[mt][nt] = __builtin_amdgcn_wmma_f32_16x16x32_bf16(
                    false, a.v, false, bfr[nt].v, (short)0, acc[mt][nt], false, false);
            }
        }
        // no second barrier: the next overwrite of buffer `cur` is issued only
        // after the next iteration's barrier, which proves all waves finished
        // reading it.
    }

    float* ob = out + (size_t)b * S_LEN * S_LEN;
#pragma unroll
    for (int mt = 0; mt < 4; ++mt) {
#pragma unroll
        for (int nt = 0; nt < 2; ++nt) {
            const int colk = k0 + kw + nt * 16 + l;
#pragma unroll
            for (int r = 0; r < 8; ++r) {
                const int rowq = q0 + qw + mt * 16 + hi * 8 + r;
                ob[(size_t)rowq * S_LEN + colk] = acc[mt][nt][r];
            }
        }
    }
}

// ---------------------------------------------------------------------------
// Kernel 3: in-place row softmax over [B*S, S]. One 256-thread block per row.
// Row (16 KB) held in registers: 16 fp32 per thread. wave32 shuffle + LDS
// cross-wave reduction.
// ---------------------------------------------------------------------------
__global__ __launch_bounds__(256) void softmax_kernel(float* __restrict__ out)
{
    __shared__ float red[8];
    const int tid = threadIdx.x;
    float* p = out + (size_t)blockIdx.x * S_LEN;

    float x[16];
#pragma unroll
    for (int i = 0; i < 16; ++i) x[i] = p[tid + i * 256];

    float m = x[0];
#pragma unroll
    for (int i = 1; i < 16; ++i) m = fmaxf(m, x[i]);
#pragma unroll
    for (int off = 16; off >= 1; off >>= 1)
        m = fmaxf(m, __shfl_xor(m, off, 32));
    if ((tid & 31) == 0) red[tid >> 5] = m;
    __syncthreads();
    m = red[0];
#pragma unroll
    for (int w = 1; w < 8; ++w) m = fmaxf(m, red[w]);
    __syncthreads();

    float s = 0.0f;
#pragma unroll
    for (int i = 0; i < 16; ++i) {
        x[i] = __expf(x[i] - m);
        s += x[i];
    }
#pragma unroll
    for (int off = 16; off >= 1; off >>= 1)
        s += __shfl_xor(s, off, 32);
    if ((tid & 31) == 0) red[tid >> 5] = s;
    __syncthreads();
    s = 0.0f;
#pragma unroll
    for (int w = 0; w < 8; ++w) s += red[w];

    const float inv = 1.0f / s;
#pragma unroll
    for (int i = 0; i < 16; ++i) p[tid + i * 256] = x[i] * inv;
}

// ---------------------------------------------------------------------------
extern "C" void kernel_launch(void* const* d_in, const int* in_sizes, int n_in,
                              void* d_out, int out_size, void* d_ws, size_t ws_size,
                              hipStream_t stream)
{
    (void)in_sizes; (void)n_in; (void)out_size; (void)ws_size;

    const float* hidden  = (const float*)d_in[0];
    const float* pre_emb = (const float*)d_in[1];
    const float* w1      = (const float*)d_in[2];
    const float* b1      = (const float*)d_in[3];
    const float* w2      = (const float*)d_in[4];
    const float* b2      = (const float*)d_in[5];
    float* out = (float*)d_out;

    bf16_t* Hws = (bf16_t*)d_ws;                                // 33.5 MB
    bf16_t* Iws = Hws + (size_t)B_SZ * S_LEN * D_MODEL;         // +33.5 MB

    dim3 g1(D_MODEL / 64, (B_SZ * S_LEN) / 64);                 // 16 x 256
    fused_linear_kernel<<<g1, 256, 0, stream>>>(hidden, pre_emb, w1, b1, w2, b2,
                                                Hws, Iws);

    dim3 g2(S_LEN / 128, S_LEN / 128, B_SZ);                    // 32 x 32 x 4
    attn_scores_kernel<<<g2, 256, 0, stream>>>(Hws, Iws, out);

    softmax_kernel<<<B_SZ * S_LEN, 256, 0, stream>>>(out);
}
